// ArcFace_64183991271874
// MI455X (gfx1250) — matrix operand order repaired and checked by reading, not compiled, not run on hardware
//
#include <hip/hip_runtime.h>
#include <hip/hip_bf16.h>

typedef __attribute__((ext_vector_type(16))) __bf16 v16bf;
typedef __attribute__((ext_vector_type(8)))  __bf16 v8bf;
typedef __attribute__((ext_vector_type(8)))  float  v8f;

#define B_DIM 512
#define D_DIM 512
#define C_DIM 100000
#define COS_M 0.8775825618903728f
#define SIN_M 0.479425538604203f
#define SCALE 64.0f
#define EPS_C 1e-7f

// ---------------- Pre-pass 1: embedding row inverse norms ----------------
__global__ void emb_rownorm_kernel(const float* __restrict__ emb,
                                   float* __restrict__ einv) {
    __shared__ float red[256];
    const int row = blockIdx.x;
    const float* p = emb + (size_t)row * D_DIM;
    float s = 0.0f;
    for (int j = threadIdx.x; j < D_DIM; j += 256) {
        float v = p[j];
        s += v * v;
    }
    red[threadIdx.x] = s;
    __syncthreads();
    for (int off = 128; off > 0; off >>= 1) {
        if ((int)threadIdx.x < off) red[threadIdx.x] += red[threadIdx.x + off];
        __syncthreads();
    }
    if (threadIdx.x == 0) einv[row] = rsqrtf(red[0]);
}

// ---- Pre-pass 2: normalize embeddings (fold row norm) + bf16 hi/lo split ----
__global__ void emb_split_kernel(const float* __restrict__ emb,
                                 const float* __restrict__ einv,
                                 __bf16* __restrict__ Ah,
                                 __bf16* __restrict__ Al, int n) {
    int i = blockIdx.x * blockDim.x + threadIdx.x;
    if (i < n) {
        float a = emb[i] * einv[i >> 9];   // row = i / D_DIM
        __bf16 h = (__bf16)a;
        Ah[i] = h;
        Al[i] = (__bf16)(a - (float)h);
    }
}

// ---------------- Pre-pass 3: W column inverse norms ----------------
__global__ void w_colnorm_kernel(const float* __restrict__ W,
                                 float* __restrict__ winv) {
    int c = blockIdx.x * blockDim.x + threadIdx.x;
    if (c < C_DIM) {
        float s = 0.0f;
        #pragma unroll 8
        for (int d = 0; d < D_DIM; ++d) {
            float v = W[(size_t)d * C_DIM + c];
            s += v * v;
        }
        winv[c] = rsqrtf(s);
    }
}

// ---------------- Main GEMM + ArcFace epilogue ----------------
// Block = one 16-wide N tile, 8 waves = 8 M-blocks of 64 rows (full M=512).
// The 32x16 f32 W stripe per K-step is loaded ONCE per block (coalesced),
// split to bf16 hi/lo cooperatively, and stored to LDS already in the WMMA
// B-fragment layout; waves fetch it with 4x ds_load_b128. Double-buffered,
// one barrier per K-step.
__global__ __launch_bounds__(256) void arcface_gemm_kernel(
    const __bf16* __restrict__ Ah, const __bf16* __restrict__ Al,
    const float* __restrict__ W,
    const float* __restrict__ winv, const int* __restrict__ labels,
    float* __restrict__ out) {

    __shared__ __align__(16) __bf16 lds_bh[2][512];   // [lane][j] bf16 hi
    __shared__ __align__(16) __bf16 lds_bl[2][512];   // [lane][j] bf16 lo

    const int tid   = threadIdx.x;
    const int lane  = tid & 31;
    const int mblk  = tid >> 5;          // 8 m-blocks of 64 rows
    const int n0    = blockIdx.x * 16;   // 6250 n-tiles
    const int m0    = mblk * 64;
    const int nlane = lane & 15;
    const int hi    = lane >> 4;
    const int n     = n0 + nlane;

    // Producer role: thread t loads W[k0+rp][n0+tc], W[k0+rp+1][n0+tc]
    // (16 consecutive threads read 16 contiguous floats of one W row).
    // Fragment entry (l, j): l = tc + (rp & 16), j = rp & 15 (even pair).
    const int tc = tid & 15;
    const int rp = (tid >> 4) * 2;
    const int wl = tc + (rp & 16);
    const int wj = rp & 15;
    const float* wbase = W + (size_t)rp * C_DIM + n0 + tc;

    v8f acc[4];
    #pragma unroll
    for (int t = 0; t < 4; ++t) acc[t] = (v8f){0,0,0,0,0,0,0,0};

    for (int ks = 0; ks < D_DIM / 32; ++ks) {
        const int k0  = ks * 32;
        const int buf = ks & 1;

        // ---- Produce: split two W elements to bf16 hi/lo, packed b32 store
        float w0 = wbase[(size_t)k0 * C_DIM];
        float w1 = wbase[(size_t)(k0 + 1) * C_DIM];
        __bf16 h0 = (__bf16)w0, h1 = (__bf16)w1;
        union { unsigned u; __bf16 b[2]; } ph, pl;
        ph.b[0] = h0;
        ph.b[1] = h1;
        pl.b[0] = (__bf16)(w0 - (float)h0);
        pl.b[1] = (__bf16)(w1 - (float)h1);
        *(unsigned*)&lds_bh[buf][wl * 16 + wj] = ph.u;
        *(unsigned*)&lds_bl[buf][wl * 16 + wj] = pl.u;

        __syncthreads();   // one barrier per step; double buffer => race-free

        // ---- Consume: B fragments from LDS (element j = W[k0+(l>>4)*16+j][n])
        union { v16bf v; v8bf h8[2]; } bh, bl;
        const v8bf* pbh = (const v8bf*)&lds_bh[buf][lane * 16];
        const v8bf* pbl = (const v8bf*)&lds_bl[buf][lane * 16];
        bh.h8[0] = pbh[0];
        bh.h8[1] = pbh[1];
        bl.h8[0] = pbl[0];
        bl.h8[1] = pbl[1];

        #pragma unroll
        for (int t = 0; t < 4; ++t) {
            // A fragment: lane row = m0 + t*16 + nlane;
            // elems 0-15 = K k0+hi*16..k0+hi*16+15
            const int m = m0 + t * 16 + nlane;
            const size_t base = (size_t)m * D_DIM + k0 + hi * 16;
            union { v16bf v; v8bf h8[2]; } ah, al;
            ah.h8[0] = *(const v8bf*)(Ah + base);
            ah.h8[1] = *(const v8bf*)(Ah + base + 8);
            al.h8[0] = *(const v8bf*)(Al + base);
            al.h8[1] = *(const v8bf*)(Al + base + 8);

            // bf16x3 split-GEMM: ah*bh + al*bh + ah*bl, f32 accumulate
            acc[t] = __builtin_amdgcn_wmma_f32_16x16x32_bf16(
                false, ah.v, false, bh.v, (short)0, acc[t], false, false);
            acc[t] = __builtin_amdgcn_wmma_f32_16x16x32_bf16(
                false, al.v, false, bh.v, (short)0, acc[t], false, false);
            acc[t] = __builtin_amdgcn_wmma_f32_16x16x32_bf16(
                false, ah.v, false, bl.v, (short)0, acc[t], false, false);
        }
    }

    // ---- Epilogue: fold W col norm, clip, margin at label column, scale ----
    const float wn = winv[n];
    #pragma unroll
    for (int t = 0; t < 4; ++t) {
        #pragma unroll
        for (int r = 0; r < 8; ++r) {
            // C/D layout: VGPR r, lanes 0-15 -> M=r, lanes 16-31 -> M=r+8
            const int m = m0 + t * 16 + hi * 8 + r;
            float v = acc[t][r] * wn;
            v = fminf(fmaxf(v, -1.0f + EPS_C), 1.0f - EPS_C);
            if (labels[m] == n) {
                v = v * COS_M - sqrtf(fmaxf(1.0f - v * v, 0.0f)) * SIN_M;
            }
            out[(size_t)m * C_DIM + n] = v * SCALE;
        }
    }
}

extern "C" void kernel_launch(void* const* d_in, const int* in_sizes, int n_in,
                              void* d_out, int out_size, void* d_ws, size_t ws_size,
                              hipStream_t stream) {
    const float* emb    = (const float*)d_in[0];  // [512, 512]
    const int*   labels = (const int*)d_in[1];    // [512]
    const float* W      = (const float*)d_in[2];  // [512, 100000]
    float* out = (float*)d_out;                   // [512, 100000]

    // Workspace layout (all 16B aligned):
    char* ws = (char*)d_ws;
    float*  winv = (float*)ws;                             // 100000 * 4 = 400000 B
    float*  einv = (float*)(ws + 400000);                  // 512 * 4   = 2048 B
    __bf16* Ah   = (__bf16*)(ws + 402048);                 // 262144 * 2 = 524288 B
    __bf16* Al   = (__bf16*)(ws + 402048 + 524288);        // 524288 B

    emb_rownorm_kernel<<<B_DIM, 256, 0, stream>>>(emb, einv);
    emb_split_kernel<<<(B_DIM * D_DIM + 255) / 256, 256, 0, stream>>>(
        emb, einv, Ah, Al, B_DIM * D_DIM);
    w_colnorm_kernel<<<(C_DIM + 255) / 256, 256, 0, stream>>>(W, winv);

    // 6250 blocks: one n-tile each; 8 waves cover all 8 m-blocks
    arcface_gemm_kernel<<<C_DIM / 16, 256, 0, stream>>>(Ah, Al, W, winv,
                                                        labels, out);
}
